// ImageCaptioning_58755152609335
// MI455X (gfx1250) — matrix operand order repaired and checked by reading, not compiled, not run on hardware
//
#include <hip/hip_runtime.h>

#define B_   64
#define H_   1024
#define V_   32000
#define T_   50
#define IOUT 1000

typedef __attribute__((ext_vector_type(16))) __bf16 v16bf;
typedef __attribute__((ext_vector_type(8)))  __bf16 v8bf;
typedef __attribute__((ext_vector_type(8)))  float  v8f;

union FragBF { v16bf v; v8bf h[2]; };

// float -> bf16, round-to-nearest-even
__device__ __forceinline__ unsigned short f2bf(float f) {
  unsigned u = __float_as_uint(f);
  unsigned r = u + 0x7FFFu + ((u >> 16) & 1u);
  return (unsigned short)(r >> 16);
}

// ---------------------------------------------------------------------------
// C[64, N] = A[64, K](bf16, row-major) * W[N, K](bf16, row-major)^T + bias[N]
// K % 32 == 0, N % 64 == 0. block = 128 threads (4 waves), grid.x = N / 64.
// Wave w handles rows m0 = 16*w; each wave accumulates NT=4 N-tiles of 16.
// Double-buffered k-loop: 10 global_load_b128 for block kb+32 are issued
// BEFORE the 4 WMMAs on block kb, so loads overlap matrix math instead of
// draining loadcnt to 0 in front of every WMMA.
// Fragment layouts used here (wave32; A and B share the contiguous k order):
//   A 16x32 bf16 : lane<16 -> K 0..15; lane>=16 -> K 16..31 (contiguous)
//   B 32x16 bf16 : lane l holds col n=(l&15), 16 contiguous K at (l>>4)*16
//   C 16x16 f32  : vgpr r -> row (l>>4)*8 + r, col (l&15)
// ---------------------------------------------------------------------------
__global__ void __launch_bounds__(128)
wmma_gemm_bf16(const unsigned short* __restrict__ A,
               const unsigned short* __restrict__ W,
               const float* __restrict__ bias,
               float* __restrict__ C, int N, int K)
{
  constexpr int NT = 4;
  const int lane = threadIdx.x & 31;
  const int wave = threadIdx.x >> 5;
  const int l15  = lane & 15;
  const int lh   = lane >> 4;          // 0 or 1 (half-wave)
  const int m0   = wave * 16;
  const int n0   = blockIdx.x * (16 * NT);

  const __bf16* a0 = (const __bf16*)(A + (size_t)(m0 + l15) * K) + lh * 16;
  const __bf16* bp[NT];
#pragma unroll
  for (int j = 0; j < NT; ++j)
    bp[j] = (const __bf16*)(W + (size_t)(n0 + j * 16 + l15) * K) + lh * 16;

  v8f acc[NT];
#pragma unroll
  for (int j = 0; j < NT; ++j) acc[j] = (v8f){};

  // ---- prologue: load k-block 0 into cur ----
  FragBF a_cur, b_cur[NT];
  a_cur.h[0] = *(const v8bf*)(a0);
  a_cur.h[1] = *(const v8bf*)(a0 + 8);
#pragma unroll
  for (int j = 0; j < NT; ++j) {
    b_cur[j].h[0] = *(const v8bf*)(bp[j]);
    b_cur[j].h[1] = *(const v8bf*)(bp[j] + 8);
  }

  // ---- pipelined main loop: prefetch kb, compute kb-32 ----
  for (int kb = 32; kb < K; kb += 32) {
    FragBF a_nxt, b_nxt[NT];
    a_nxt.h[0] = *(const v8bf*)(a0 + kb);
    a_nxt.h[1] = *(const v8bf*)(a0 + kb + 8);
#pragma unroll
    for (int j = 0; j < NT; ++j) {
      b_nxt[j].h[0] = *(const v8bf*)(bp[j] + kb);
      b_nxt[j].h[1] = *(const v8bf*)(bp[j] + kb + 8);
    }
#pragma unroll
    for (int j = 0; j < NT; ++j)
      acc[j] = __builtin_amdgcn_wmma_f32_16x16x32_bf16(
          false, a_cur.v, false, b_cur[j].v, (short)0, acc[j], false, false);
    a_cur = a_nxt;
#pragma unroll
    for (int j = 0; j < NT; ++j) b_cur[j] = b_nxt[j];
  }

  // ---- epilogue: last k-block ----
#pragma unroll
  for (int j = 0; j < NT; ++j)
    acc[j] = __builtin_amdgcn_wmma_f32_16x16x32_bf16(
        false, a_cur.v, false, b_cur[j].v, (short)0, acc[j], false, false);

  const int row0 = m0 + lh * 8;
#pragma unroll
  for (int j = 0; j < NT; ++j) {
    const int col = n0 + j * 16 + l15;
    const float bb = bias[col];
#pragma unroll
    for (int r = 0; r < 8; ++r)
      C[(size_t)(row0 + r) * N + col] = acc[j][r] + bb;
  }
}

// ---------------------------------------------------------------------------
// One-time weight conversion f32 -> bf16 (grid-stride)
// ---------------------------------------------------------------------------
__global__ void cvt_bf16(const float* __restrict__ src,
                         unsigned short* __restrict__ dst, int n)
{
  int stride = gridDim.x * blockDim.x;
  for (int i = blockIdx.x * blockDim.x + threadIdx.x; i < n; i += stride)
    dst[i] = f2bf(src[i]);
}

// ---------------------------------------------------------------------------
// h0 = img_feats @ fcn_w^T + fcn_b   (one-time, tiny: K=1000 not WMMA-friendly)
// ---------------------------------------------------------------------------
__global__ void init_h(const float* __restrict__ img, const float* __restrict__ fw,
                       const float* __restrict__ fb, float* __restrict__ h,
                       unsigned short* __restrict__ hbf)
{
  int idx = blockIdx.x * blockDim.x + threadIdx.x;   // < B_*H_
  int b = idx >> 10;
  int j = idx & (H_ - 1);
  float s = fb[j];
  const float* ip = img + (size_t)b * IOUT;
  const float* wp = fw  + (size_t)j * IOUT;
  for (int k = 0; k < IOUT; ++k) s += ip[k] * wp[k];
  h[idx]   = s;
  hbf[idx] = f2bf(s);
}

// ---------------------------------------------------------------------------
// x = relu(emb[token]) -> bf16. token = SOS(0) at t=0 else sentences[b][t-1]
// ---------------------------------------------------------------------------
__global__ void embed_relu(const int* __restrict__ sent, int t,
                           const float* __restrict__ emb,
                           unsigned short* __restrict__ xbf)
{
  int idx = blockIdx.x * blockDim.x + threadIdx.x;   // < B_*H_
  int b = idx >> 10;
  int j = idx & (H_ - 1);
  int tok = (t == 0) ? 0 : sent[b * T_ + (t - 1)];
  float v = emb[(size_t)tok * H_ + j];
  xbf[idx] = f2bf(v > 0.f ? v : 0.f);
}

// ---------------------------------------------------------------------------
// GRU gate fusion: gi/gh are [B, 3H] (bias already folded in by GEMM).
// Updates h (f32, in place) and hbf (bf16) for the next step's GEMMs.
// ---------------------------------------------------------------------------
__global__ void gru_gate(const float* __restrict__ gi, const float* __restrict__ gh,
                         float* __restrict__ h, unsigned short* __restrict__ hbf)
{
  int idx = blockIdx.x * blockDim.x + threadIdx.x;   // < B_*H_
  int b = idx >> 10;
  int j = idx & (H_ - 1);
  size_t base = (size_t)b * 3 * H_ + j;
  float r = gi[base]          + gh[base];
  float z = gi[base +     H_] + gh[base +     H_];
  r = 1.f / (1.f + expf(-r));
  z = 1.f / (1.f + expf(-z));
  float n = tanhf(gi[base + 2 * H_] + r * gh[base + 2 * H_]);
  float hn = (1.f - z) * n + z * h[idx];
  h[idx]   = hn;
  hbf[idx] = f2bf(hn);
}

// ---------------------------------------------------------------------------
// In-place row-wise log-softmax over V elements. grid.x = rows, block = 512.
// ---------------------------------------------------------------------------
__global__ void log_softmax_rows(float* __restrict__ data, int V)
{
  __shared__ float red[32];
  float* p = data + (size_t)blockIdx.x * V;
  const int tid = threadIdx.x;
  const int nw  = blockDim.x >> 5;

  // pass 1: max
  float m = -3.402823466e38f;
  for (int i = tid; i < V; i += blockDim.x) m = fmaxf(m, p[i]);
  for (int o = 16; o > 0; o >>= 1) m = fmaxf(m, __shfl_xor(m, o, 32));
  if ((tid & 31) == 0) red[tid >> 5] = m;
  __syncthreads();
  if (tid < 32) {
    float v = (tid < nw) ? red[tid] : -3.402823466e38f;
    for (int o = 16; o > 0; o >>= 1) v = fmaxf(v, __shfl_xor(v, o, 32));
    if (tid == 0) red[0] = v;
  }
  __syncthreads();
  m = red[0];
  __syncthreads();

  // pass 2: sum of exp
  float s = 0.f;
  for (int i = tid; i < V; i += blockDim.x) s += expf(p[i] - m);
  for (int o = 16; o > 0; o >>= 1) s += __shfl_xor(s, o, 32);
  if ((tid & 31) == 0) red[tid >> 5] = s;
  __syncthreads();
  if (tid < 32) {
    float v = (tid < nw) ? red[tid] : 0.f;
    for (int o = 16; o > 0; o >>= 1) v += __shfl_xor(v, o, 32);
    if (tid == 0) red[0] = v;
  }
  __syncthreads();
  const float lse = m + logf(red[0]);

  // pass 3: write log-probs
  for (int i = tid; i < V; i += blockDim.x) p[i] -= lse;
}

// ---------------------------------------------------------------------------
extern "C" void kernel_launch(void* const* d_in, const int* in_sizes, int n_in,
                              void* d_out, int out_size, void* d_ws, size_t ws_size,
                              hipStream_t stream)
{
  const float* img   = (const float*)d_in[0];
  const int*   sent  = (const int*)  d_in[1];
  const float* fcn_w = (const float*)d_in[2];
  const float* fcn_b = (const float*)d_in[3];
  const float* emb   = (const float*)d_in[4];
  const float* W_ih  = (const float*)d_in[5];
  const float* W_hh  = (const float*)d_in[6];
  const float* b_ih  = (const float*)d_in[7];
  const float* b_hh  = (const float*)d_in[8];
  const float* out_w = (const float*)d_in[9];
  const float* out_b = (const float*)d_in[10];
  (void)in_sizes; (void)n_in; (void)out_size; (void)ws_size;

  // Workspace carve-up (~80 MB), all chunks 256B-aligned by construction.
  char* w = (char*)d_ws;
  unsigned short* outw_bf = (unsigned short*)w; w += (size_t)V_ * H_ * 2;      // 65.5 MB
  unsigned short* wih_bf  = (unsigned short*)w; w += (size_t)3 * H_ * H_ * 2;  // 6.3 MB
  unsigned short* whh_bf  = (unsigned short*)w; w += (size_t)3 * H_ * H_ * 2;  // 6.3 MB
  float*          h       = (float*)w;          w += (size_t)B_ * H_ * 4;
  unsigned short* hbf     = (unsigned short*)w; w += (size_t)B_ * H_ * 2;
  unsigned short* xbf     = (unsigned short*)w; w += (size_t)B_ * H_ * 2;
  float*          gi      = (float*)w;          w += (size_t)B_ * 3 * H_ * 4;
  float*          gh      = (float*)w;          w += (size_t)B_ * 3 * H_ * 4;

  // One-time (per launch, deterministic): bf16 weight images + h0.
  cvt_bf16<<<4096, 256, 0, stream>>>(out_w, outw_bf, V_ * H_);
  cvt_bf16<<<1024, 256, 0, stream>>>(W_ih,  wih_bf,  3 * H_ * H_);
  cvt_bf16<<<1024, 256, 0, stream>>>(W_hh,  whh_bf,  3 * H_ * H_);
  init_h<<<(B_ * H_) / 256, 256, 0, stream>>>(img, fcn_w, fcn_b, h, hbf);

  float* out = (float*)d_out;
  for (int t = 0; t < T_; ++t) {
    embed_relu<<<(B_ * H_) / 256, 256, 0, stream>>>(sent, t, emb, xbf);
    // gi = relu(emb) @ W_ih^T + b_ih      [64, 3072]
    wmma_gemm_bf16<<<(3 * H_) / 64, 128, 0, stream>>>(xbf, wih_bf, b_ih, gi, 3 * H_, H_);
    // gh = h @ W_hh^T + b_hh              [64, 3072]
    wmma_gemm_bf16<<<(3 * H_) / 64, 128, 0, stream>>>(hbf, whh_bf, b_hh, gh, 3 * H_, H_);
    gru_gate<<<(B_ * H_) / 256, 256, 0, stream>>>(gi, gh, h, hbf);
    // logits = h_new @ out_w^T + out_b    [64, 32000] -> directly into d_out[t]
    float* logits = out + (size_t)t * B_ * V_;
    wmma_gemm_bf16<<<V_ / 64, 128, 0, stream>>>(hbf, outw_bf, out_b, logits, V_, H_);
    log_softmax_rows<<<B_, 512, 0, stream>>>(logits, V_);
  }
}
